// GlobalSelfAttention_13348758356079
// MI455X (gfx1250) — matrix operand order repaired and checked by reading, not compiled, not run on hardware
//
#include <hip/hip_runtime.h>

typedef __attribute__((ext_vector_type(16))) _Float16 v16h;
typedef __attribute__((ext_vector_type(8)))  _Float16 h8;
typedef __attribute__((ext_vector_type(2)))  _Float16 h2;
typedef __attribute__((ext_vector_type(8)))  float    v8f;
typedef __attribute__((ext_vector_type(4)))  int      i4;

static constexpr int CH     = 64;
static constexpr int NPIX   = 9216;          // 96*96
static constexpr int BATCH  = 2;
static constexpr int JSPLIT = 4;             // key-range split for occupancy
static constexpr int JCHUNK = NPIX / JSPLIT; // 2304
static constexpr int NSTEPS = JCHUNK / 32;   // 72 j-slabs per block
#define LOG2E 1.4426950408889634f

#if __has_builtin(__builtin_amdgcn_global_load_async_to_lds_b128)
#define HAVE_ASYNC_LDS 1
#else
#define HAVE_ASYNC_LDS 0
#endif

__device__ __forceinline__ float fast_exp2(float x) {
#if __has_builtin(__builtin_amdgcn_exp2f)
    return __builtin_amdgcn_exp2f(x);
#else
    return exp2f(x);
#endif
}

// 16B global -> LDS copy, per lane. Async (ASYNCcnt) when available.
__device__ __forceinline__ void copy16(const _Float16* g, _Float16* l) {
#if HAVE_ASYNC_LDS
    __builtin_amdgcn_global_load_async_to_lds_b128(
        (__attribute__((address_space(1))) i4*)(g),
        (__attribute__((address_space(3))) i4*)(l), 0, 0);
#else
    *(h8*)l = *(const h8*)g;
#endif
}

__device__ __forceinline__ void stage_wait() {
#if HAVE_ASYNC_LDS
#if __has_builtin(__builtin_amdgcn_s_wait_asynccnt)
    __builtin_amdgcn_s_wait_asynccnt(0);
#else
    asm volatile("s_wait_asynccnt 0" ::: "memory");
#endif
#else
    asm volatile("s_wait_dscnt 0" ::: "memory");
#endif
}

// 16-bit WMMA A-fragment (16 x K=32 slice): elem h <-> K = (h&7)+(h>=8?16:0)+laneOff.
// p already includes per-lane offset; elems 0-7 = p[0..7], 8-15 = p[16..23].
__device__ __forceinline__ v16h load_afrag(const _Float16* p) {
    h8 a = *(const h8*)(p);
    h8 b = *(const h8*)(p + 16);
    v16h r;
#pragma unroll
    for (int t = 0; t < 8; ++t) { r[t] = a[t]; r[t + 8] = b[t]; }
    return r;
}

// Cooperative staging of one 32-key slab: K^T tile (32 x 64c, contiguous 4KB)
// and V tile (64c x 32j, 64B rows). 128 threads x 4 chunks of 16B.
__device__ __forceinline__ void stage_tiles(const _Float16* __restrict__ kt,
                                            const _Float16* __restrict__ vm,
                                            _Float16* kb, _Float16* vb,
                                            int b, int jb, int tid)
{
    const _Float16* ks = kt + ((size_t)b * NPIX + jb) * CH;
#pragma unroll
    for (int c = 0; c < 2; ++c) {
        const int chunk = tid + c * 128;            // 0..255
        copy16(ks + chunk * 8, kb + chunk * 8);
    }
#pragma unroll
    for (int c = 0; c < 2; ++c) {
        const int chunk = tid + c * 128;            // 0..255
        const int row = chunk >> 2;                 // c channel
        const int col = (chunk & 3) * 8;            // j offset
        copy16(vm + ((size_t)b * CH + row) * NPIX + jb + col, vb + row * 32 + col);
    }
}

// ---------------------------------------------------------------------------
// Projections (fp32 math). q is pre-scaled by log2(e) so the softmax runs in
// log2 domain (bare v_exp_f32). Layouts: qt,kt: [B][N][C] f16, v: [B][C][N].
// ---------------------------------------------------------------------------
__global__ __launch_bounds__(256) void proj_kernel(
    const float* __restrict__ x,
    const float* __restrict__ Wq, const float* __restrict__ bq,
    const float* __restrict__ Wk, const float* __restrict__ bk,
    const float* __restrict__ Wv, const float* __restrict__ bv,
    _Float16* __restrict__ qt, _Float16* __restrict__ kt,
    _Float16* __restrict__ vm)
{
    const int n = blockIdx.x * blockDim.x + threadIdx.x;
    const int b = blockIdx.y;

    float xi[CH];
#pragma unroll
    for (int i = 0; i < CH; ++i)
        xi[i] = x[((size_t)b * CH + i) * NPIX + n];

    const size_t qkrow = ((size_t)b * NPIX + n) * CH;
    for (int o = 0; o < CH; ++o) {
        float aq = bq[o], ak = bk[o], av = bv[o];
        const float* wq = Wq + o * CH;
        const float* wk = Wk + o * CH;
        const float* wv = Wv + o * CH;
#pragma unroll
        for (int i = 0; i < CH; ++i) {
            const float xv = xi[i];
            aq = fmaf(wq[i], xv, aq);
            ak = fmaf(wk[i], xv, ak);
            av = fmaf(wv[i], xv, av);
        }
        qt[qkrow + o] = (_Float16)(aq * LOG2E);
        kt[qkrow + o] = (_Float16)ak;
        vm[((size_t)b * CH + o) * NPIX + n] = (_Float16)av;
    }
}

// ---------------------------------------------------------------------------
// Flash attention over one key-range chunk. 4 waves/block, each owns 32 query
// rows (two 16-row M-tiles). K/V slabs double-buffered in LDS via async
// copies. Emits unnormalized partials (O, m, l) for the combine pass.
// ---------------------------------------------------------------------------
__global__ __launch_bounds__(128) void attn_kernel(
    const _Float16* __restrict__ qt,
    const _Float16* __restrict__ kt,
    const _Float16* __restrict__ vm,
    float* __restrict__ Opart, float* __restrict__ mpart,
    float* __restrict__ lpart)
{
    __shared__ _Float16 kbuf[2][32][CH];     // 2 x 4KB
    __shared__ _Float16 vbuf[2][CH][32];     // 2 x 4KB
    __shared__ _Float16 pbuf[4][2][16][32];  // 8KB, per-wave P bounce

    const int tid  = threadIdx.x;
    const int lane = tid & 31;
    const int wid  = tid >> 5;
    const int js   = blockIdx.y;
    const int b    = blockIdx.z;
    const int i0   = (blockIdx.x * 4 + wid) * 32;
    const int li   = lane & 15;
    const int hi   = lane >> 4;
    const int jlo  = js * JCHUNK;

    // Q as B-fragments (B elem h <-> K = (h&7)+(h>=8?16:0)+8*hi, same as A)
    v16h bq[2][2];
#pragma unroll
    for (int t = 0; t < 2; ++t) {
        const _Float16* qrow =
            qt + ((size_t)b * NPIX + (i0 + t * 16 + li)) * CH + hi * 8;
        bq[t][0] = load_afrag(qrow);
        bq[t][1] = load_afrag(qrow + 32);
    }

    v8f   O[2][4] = {};
    float m[2]    = { -__builtin_inff(), -__builtin_inff() };
    float lsum[2] = { 0.f, 0.f };

    stage_tiles(kt, vm, &kbuf[0][0][0], &vbuf[0][0][0], b, jlo, tid);

    for (int s = 0; s < NSTEPS; ++s) {
        const int cur = s & 1;
        stage_wait();
        __syncthreads();
        if (s + 1 < NSTEPS)
            stage_tiles(kt, vm, &kbuf[cur ^ 1][0][0], &vbuf[cur ^ 1][0][0],
                        b, jlo + (s + 1) * 32, tid);

        // K^T A-fragments from LDS
        const _Float16* krow = &kbuf[cur][li][hi * 8];
        const v16h ak0l = load_afrag(krow);
        const v16h ak0h = load_afrag(krow + 32);
        const v16h ak1l = load_afrag(krow + 16 * CH);
        const v16h ak1h = load_afrag(krow + 16 * CH + 32);

#pragma unroll
        for (int t = 0; t < 2; ++t) {
            const v8f z = {};
            v8f s0 = __builtin_amdgcn_wmma_f32_16x16x32_f16(
                false, ak0l, false, bq[t][0], (short)0, z, false, false);
            s0 = __builtin_amdgcn_wmma_f32_16x16x32_f16(
                false, ak0h, false, bq[t][1], (short)0, s0, false, false);
            v8f s1 = __builtin_amdgcn_wmma_f32_16x16x32_f16(
                false, ak1l, false, bq[t][0], (short)0, z, false, false);
            s1 = __builtin_amdgcn_wmma_f32_16x16x32_f16(
                false, ak1h, false, bq[t][1], (short)0, s1, false, false);

            float mx = s0[0];
#pragma unroll
            for (int r = 1; r < 8; ++r) mx = fmaxf(mx, s0[r]);
#pragma unroll
            for (int r = 0; r < 8; ++r) mx = fmaxf(mx, s1[r]);
            mx = fmaxf(mx, __shfl_xor(mx, 16, 32));

            const float mold = m[t];
            const float mnew = fmaxf(mold, mx);
            m[t] = mnew;
            if (__any(mnew > mold)) {        // wave-uniform: skip when max unchanged
                const float scale = fast_exp2(mold - mnew);
                lsum[t] *= scale;
#pragma unroll
                for (int cb = 0; cb < 4; ++cb)
#pragma unroll
                    for (int r = 0; r < 8; ++r) O[t][cb][r] *= scale;
            }

            float ls = 0.f;
            union { h8 v; h2 p[4]; } ph0, ph1;
#pragma unroll
            for (int r = 0; r < 4; ++r) {
                const float a = fast_exp2(s0[2 * r]     - mnew);
                const float c = fast_exp2(s0[2 * r + 1] - mnew);
                const float d = fast_exp2(s1[2 * r]     - mnew);
                const float e = fast_exp2(s1[2 * r + 1] - mnew);
                ls += (a + c) + (d + e);
                ph0.p[r] = (h2)__builtin_amdgcn_cvt_pkrtz(a, c);
                ph1.p[r] = (h2)__builtin_amdgcn_cvt_pkrtz(d, e);
            }
            ls += __shfl_xor(ls, 16, 32);
            lsum[t] += ls;

            *(h8*)&pbuf[wid][t][li][hi * 8]      = ph0.v;
            *(h8*)&pbuf[wid][t][li][16 + hi * 8] = ph1.v;
        }

        asm volatile("s_wait_dscnt 0" ::: "memory");

        v16h bp[2];
#pragma unroll
        for (int t = 0; t < 2; ++t)
            bp[t] = load_afrag(&pbuf[wid][t][li][hi * 8]);

#pragma unroll
        for (int cb = 0; cb < 4; ++cb) {
            const v16h av = load_afrag(&vbuf[cur][cb * 16 + li][hi * 8]);
#pragma unroll
            for (int t = 0; t < 2; ++t)
                O[t][cb] = __builtin_amdgcn_wmma_f32_16x16x32_f16(
                    false, av, false, bp[t], (short)0, O[t][cb], false, false);
        }
    }

    // write unnormalized partials
    float* ob = Opart + ((size_t)(js * BATCH + b) * CH) * NPIX;
#pragma unroll
    for (int t = 0; t < 2; ++t) {
        const int ig = i0 + t * 16 + li;
#pragma unroll
        for (int cb = 0; cb < 4; ++cb)
#pragma unroll
            for (int r = 0; r < 8; ++r) {
                const int c = cb * 16 + hi * 8 + r;
                ob[(size_t)c * NPIX + ig] = O[t][cb][r];
            }
        if (lane < 16) {
            mpart[(size_t)(js * BATCH + b) * NPIX + ig] = m[t];
            lpart[(size_t)(js * BATCH + b) * NPIX + ig] = lsum[t];
        }
    }
}

// ---------------------------------------------------------------------------
// Merge the JSPLIT partials:  out = sum_s O_s*2^(m_s-m) / sum_s l_s*2^(m_s-m)
// ---------------------------------------------------------------------------
__global__ __launch_bounds__(256) void combine_kernel(
    const float* __restrict__ Opart, const float* __restrict__ mpart,
    const float* __restrict__ lpart, float* __restrict__ out)
{
    const int idx = blockIdx.x * 256 + threadIdx.x;   // over B*CH*NPIX
    const int i   = idx % NPIX;
    const int bc  = idx / NPIX;                       // b*CH + c
    const int b   = bc / CH;

    float mg = -__builtin_inff();
#pragma unroll
    for (int s = 0; s < JSPLIT; ++s)
        mg = fmaxf(mg, mpart[(size_t)(s * BATCH + b) * NPIX + i]);

    float l = 0.f, o = 0.f;
#pragma unroll
    for (int s = 0; s < JSPLIT; ++s) {
        const float w = fast_exp2(mpart[(size_t)(s * BATCH + b) * NPIX + i] - mg);
        l = fmaf(lpart[(size_t)(s * BATCH + b) * NPIX + i], w, l);
        o = fmaf(Opart[((size_t)s * BATCH * CH + bc) * NPIX + i], w, o);
    }
    out[(size_t)bc * NPIX + i] = o / l;
}

extern "C" void kernel_launch(void* const* d_in, const int* in_sizes, int n_in,
                              void* d_out, int out_size, void* d_ws, size_t ws_size,
                              hipStream_t stream) {
    const float* x  = (const float*)d_in[0];
    const float* Wq = (const float*)d_in[1];
    const float* bq = (const float*)d_in[2];
    const float* Wk = (const float*)d_in[3];
    const float* bk = (const float*)d_in[4];
    const float* Wv = (const float*)d_in[5];
    const float* bv = (const float*)d_in[6];
    float* out = (float*)d_out;

    const size_t nhalf = (size_t)BATCH * NPIX * CH;   // per f16 matrix
    _Float16* qt = (_Float16*)d_ws;
    _Float16* kt = qt + nhalf;
    _Float16* vm = kt + nhalf;
    float* Opart = (float*)(vm + nhalf);                             // [S][B][C][N]
    float* mpart = Opart + (size_t)JSPLIT * BATCH * CH * NPIX;       // [S][B][N]
    float* lpart = mpart + (size_t)JSPLIT * BATCH * NPIX;

    dim3 pg(NPIX / 256, BATCH);
    proj_kernel<<<pg, 256, 0, stream>>>(x, Wq, bq, Wk, bk, Wv, bv, qt, kt, vm);

    dim3 ag(NPIX / 128, JSPLIT, BATCH);   // 576 blocks x 4 waves
    attn_kernel<<<ag, 128, 0, stream>>>(qt, kt, vm, Opart, mpart, lpart);

    dim3 cg((size_t)BATCH * CH * NPIX / 256);
    combine_kernel<<<cg, 256, 0, stream>>>(Opart, mpart, lpart, out);
}
